// DRHG_MoE_51762945851654
// MI455X (gfx1250) — compile-verified, matrix operand order checked by reading, not run on hardware
//
#include <hip/hip_runtime.h>

// ---------------------------------------------------------------------------
// Types for CDNA5 WMMA (wave32, 16x16x32 bf16 -> f32)
// ---------------------------------------------------------------------------
typedef __bf16 bf16;
typedef __attribute__((ext_vector_type(16))) __bf16 v16bf;
typedef __attribute__((ext_vector_type(8)))  __bf16 v8bf;
typedef __attribute__((ext_vector_type(8)))  float  v8f;

__device__ __forceinline__ v8f wmma_bf16(v16bf a, v16bf b, v8f c) {
  return __builtin_amdgcn_wmma_f32_16x16x32_bf16(false, a, false, b, (short)0, c,
                                                 false, false);
}

// A-operand fragment (16-bit A 16x32 layout, ISA 7.12.2):
// lane l: row = l&15, kbase = (l>>4)*8; elems 0..7 -> K=kbase..kbase+7 (contig),
// elems 8..15 -> K=kbase+16..kbase+23 (contig). Two 16-byte loads.
// B fragment loaded identically from the transposed (N x K) weight layout.
__device__ __forceinline__ v16bf load_frag(const bf16* p) {
  v8bf lo = *(const v8bf*)(p);
  v8bf hi = *(const v8bf*)(p + 16);
  v16bf f;
#pragma unroll
  for (int i = 0; i < 8; ++i) { f[i] = lo[i]; f[i + 8] = hi[i]; }
  return f;
}

// ---------------------------------------------------------------------------
// Problem constants
// ---------------------------------------------------------------------------
#define NB    1024        // batch
#define XD    62
#define XDP   64          // padded channel dim
#define TW    10
#define INCH  128
#define MID   256
#define OUTD  512
#define K1D   10
#define K1P   16
#define LIN   512
#define KREG  1152        // 9*128 padded region K
#define KCAT1 384         // 3*128
#define KCAT2 768         // 3*256
#define FEAT  36864       // 72*512
#define N3P   64          // head-3 output padded 4 -> 64 (guard-free GEMM)

__constant__ int c_region_idx[10][9] = {
  { 0, 1, 2, 3, 4,-1,-1,-1,-1},
  { 5, 6, 7,14,15,16,-1,-1,-1},
  { 8, 9,10,17,18,19,-1,-1,-1},
  {11,12,13,20,21,22,-1,-1,-1},
  {23,24,25,32,33,34,-1,-1,-1},
  {26,27,28,35,36,37,-1,-1,-1},
  {29,30,31,38,39,40,-1,-1,-1},
  {41,42,43,50,51,57,-1,-1,-1},
  {44,45,46,52,53,54,58,59,60},
  {47,48,49,55,56,61,-1,-1,-1}};
__constant__ int c_region_len[10] = {5,6,6,6,6,6,6,6,9,6};

// ---------------------------------------------------------------------------
// Tiled bf16 GEMM:  C(MxN) = A(MxK, row-major) * Bt(NxK, row-major)^T
// Requirements (all launches satisfy them): M % 128 == 0, N % 64 == 0,
// K % 32 == 0, lda/ldb multiples of 8 (16B-aligned rows).
// 256 threads = 8 waves; 128x64 block tile; each wave owns a 32x32 patch
// (2x2 WMMA sub-tiles) -> 4 v_wmma per k-step, fragments reused 2x.
// grid.z either batches (pointer strides) or splits K (partial C buffers).
// ---------------------------------------------------------------------------
#define BM 128
#define BN 64
#define LPITCH 48   // LDS row pitch in bf16 (96B, 16B aligned, bank-spread)

__global__ __launch_bounds__(256) void gemm_bf16_kernel(
    const bf16* __restrict__ A, long lda, long sAz,
    const bf16* __restrict__ Bt, long ldb, long sBz,
    float* __restrict__ C, long ldc, long sCz,
    long M, int K,
    const float* __restrict__ bias, long sbz,
    int relu, int ksplit)
{
  __shared__ bf16 As[BM * LPITCH];
  __shared__ bf16 Bs[BN * LPITCH];

  const int z = blockIdx.z;
  int kbeg = 0, kend = K;
  if (ksplit > 1) {
    const int kc = K / ksplit;
    kbeg = z * kc; kend = kbeg + kc;
    C += (long)z * M * ldc;      // partial buffer; epilogue deferred
    bias = nullptr; relu = 0;
  } else {
    A += (long)z * sAz; Bt += (long)z * sBz; C += (long)z * sCz;
    if (bias) bias += (long)z * sbz;
  }

  const int tid  = threadIdx.x;
  const int wave = tid >> 5, lane = tid & 31;
  const long m0 = (long)blockIdx.x * BM;
  const long n0 = (long)blockIdx.y * BN;
  const int wm = (wave & 3) * 32;       // 2 M sub-tiles: wm, wm+16
  const int wn = (wave >> 2) * 32;      // 2 N sub-tiles: wn, wn+16
  const int fr = lane & 15, fk = (lane >> 4) * 8;

  // staging: A = 128 rows x 32 halves (2 x uint4 per thread),
  //          B =  64 rows x 32 halves (1 x uint4 per thread)
  const int arow = tid >> 1;            // 0..127
  const int ak   = (tid & 1) * 16;      // 0,16
  const int brow = tid >> 2;            // 0..63
  const int bk   = (tid & 3) * 8;       // 0,8,16,24

  const bf16* pA = A + (m0 + arow) * lda + kbeg + ak;
  const bf16* pB = Bt + (n0 + brow) * ldb + kbeg + bk;

  v8f acc00 = {}, acc01 = {}, acc10 = {}, acc11 = {};

  for (int k0 = kbeg; k0 < kend; k0 += 32) {
    const uint4 da0 = *(const uint4*)(pA);
    const uint4 da1 = *(const uint4*)(pA + 8);
    const uint4 db  = *(const uint4*)(pB);
    // Unconditional speculative prefetch of the next k-tile (tail prefetch
    // past the operand is harmless: dropped on translation failure, and the
    // workspace slab extends past every operand). Keeps the loop branch-free.
    __builtin_prefetch(pA + 32, 0, 1);
    __builtin_prefetch(pB + 32, 0, 1);
    pA += 32; pB += 32;
    *(uint4*)(&As[arow * LPITCH + ak])     = da0;
    *(uint4*)(&As[arow * LPITCH + ak + 8]) = da1;
    *(uint4*)(&Bs[brow * LPITCH + bk])     = db;
    __syncthreads();

    v16bf a0 = load_frag(&As[(wm + fr) * LPITCH + fk]);
    v16bf a1 = load_frag(&As[(wm + 16 + fr) * LPITCH + fk]);
    v16bf b0 = load_frag(&Bs[(wn + fr) * LPITCH + fk]);
    v16bf b1 = load_frag(&Bs[(wn + 16 + fr) * LPITCH + fk]);
    acc00 = wmma_bf16(a0, b0, acc00);
    acc01 = wmma_bf16(a0, b1, acc01);
    acc10 = wmma_bf16(a1, b0, acc10);
    acc11 = wmma_bf16(a1, b1, acc11);
    __syncthreads();
  }

  // D layout: VGPR r -> row (r + 8*(lane>=16)), col = lane&15  (ISA 7.12.2)
  const int colA  = (lane & 15);
  const int rsub  = (lane >> 4) * 8;
  const long row0 = m0 + wm + rsub;
  const long col0 = n0 + wn + colA;
  const float bi0 = bias ? bias[col0] : 0.f;
  const float bi1 = bias ? bias[col0 + 16] : 0.f;
#pragma unroll
  for (int r = 0; r < 8; ++r) {
    float v00 = acc00[r] + bi0, v01 = acc01[r] + bi1;
    float v10 = acc10[r] + bi0, v11 = acc11[r] + bi1;
    if (relu) {
      v00 = fmaxf(v00, 0.f); v01 = fmaxf(v01, 0.f);
      v10 = fmaxf(v10, 0.f); v11 = fmaxf(v11, 0.f);
    }
    C[(row0 + r) * ldc + col0]           = v00;
    C[(row0 + r) * ldc + col0 + 16]      = v01;
    C[(row0 + 16 + r) * ldc + col0]      = v10;
    C[(row0 + 16 + r) * ldc + col0 + 16] = v11;
  }
}

__global__ void ksplit_reduce_kernel(const float* __restrict__ Cpart,
                                     float* __restrict__ C,
                                     const float* __restrict__ bias,
                                     long elems, int cols, int ks)
{
  long idx = (long)blockIdx.x * blockDim.x + threadIdx.x;
  if (idx >= elems) return;
  float s = bias ? bias[idx % cols] : 0.f;
  for (int zz = 0; zz < ks; ++zz) s += Cpart[(long)zz * elems + idx];
  C[idx] = s;
}

// ---------------------------------------------------------------------------
// Stage 1: g[b,c,t] = (sum_i x*tfe_w[c,i] + tfe_b[c]) * exp(-var_i(x, ddof=1))
// Writes bf16 into Gcat cols [0,128) (zero pad rows 62..63) and gT (t-major).
// ---------------------------------------------------------------------------
__global__ void preproc_kernel(const float* __restrict__ x,
                               const float* __restrict__ tfe_w,
                               const float* __restrict__ tfe_b,
                               bf16* __restrict__ Gcat,   // [B][64][384]
                               bf16* __restrict__ gT)     // [B][128][64]
{
  long idx = (long)blockIdx.x * blockDim.x + threadIdx.x;  // B*64*128
  const int  t = (int)(idx & 127);
  const int  c = (int)((idx >> 7) & 63);
  const long b = idx >> 13;
  float g = 0.f;
  if (c < XD) {
    const float* xp = x + ((b * XD + c) * TW) * INCH + t;
    float s1 = 0.f, s2 = 0.f, gd = 0.f;
#pragma unroll
    for (int i = 0; i < TW; ++i) {
      const float v = xp[i * INCH];
      s1 += v; s2 += v * v;
      gd += v * tfe_w[c * TW + i];
    }
    const float mean = s1 * 0.1f;
    const float var  = (s2 - 10.f * mean * mean) * (1.f / 9.f);  // ddof=1
    g = (gd + tfe_b[c]) * __expf(-var);
  }
  Gcat[(b * XDP + c) * KCAT1 + t] = (bf16)g;
  gT[(b * INCH + t) * XDP + c]    = (bf16)g;
}

// ---------------------------------------------------------------------------
// Small dense helpers (A^2, A1^2, weight packing to Bt/bf16)
// ---------------------------------------------------------------------------
__global__ void matmul62_kernel(const float* __restrict__ A, float* __restrict__ A2)
{
  int idx = blockIdx.x * blockDim.x + threadIdx.x;
  if (idx >= XD * XD) return;
  const int n = idx / XD, m = idx % XD;
  float s = 0.f;
  for (int k = 0; k < XD; ++k) s += A[n * XD + k] * A[k * XD + m];
  A2[idx] = s;
}

__global__ void pack_S_kernel(const float* __restrict__ A,
                              const float* __restrict__ A2,
                              bf16* __restrict__ Spad)   // [2][64][64]
{
  int idx = blockIdx.x * blockDim.x + threadIdx.x;
  if (idx >= 2 * 64 * 64) return;
  const int sel = idx / 4096, rc = idx % 4096, rr = rc >> 6, cc = rc & 63;
  const float* src = sel ? A2 : A;
  Spad[idx] = (bf16)((rr < XD && cc < XD) ? src[rr * XD + cc] : 0.f);
}

__global__ void matmul10_kernel(const float* __restrict__ A1, float* __restrict__ A1sq)
{
  int idx = blockIdx.x * blockDim.x + threadIdx.x;
  if (idx >= 100) return;
  const int n = idx / 10, m = idx % 10;
  float s = 0.f;
  for (int k = 0; k < 10; ++k) s += A1[n * 10 + k] * A1[k * 10 + m];
  A1sq[idx] = s;
}

__global__ void pack_S1_kernel(const float* __restrict__ A1,
                               const float* __restrict__ A1sq,
                               bf16* __restrict__ S1pad)  // [2][16][32]
{
  int idx = blockIdx.x * blockDim.x + threadIdx.x;
  if (idx >= 2 * 16 * 32) return;
  const int sel = idx / 512, rc = idx % 512, rr = rc >> 5, cc = rc & 31;
  const float* src = sel ? A1sq : A1;
  S1pad[idx] = (bf16)((rr < 10 && cc < 10) ? src[rr * 10 + cc] : 0.f);
}

// cheby1_w (3,128,512) -> Wt [512][384], Wt[o][k*128+t]
__global__ void pack_wcat1_kernel(const float* __restrict__ w, bf16* __restrict__ Wt)
{
  int idx = blockIdx.x * blockDim.x + threadIdx.x;
  if (idx >= OUTD * KCAT1) return;
  const int kk = idx % KCAT1, o = idx / KCAT1;
  Wt[idx] = (bf16)w[kk * OUTD + o];   // (k*128+t)*512 + o
}

// cheby2_w (3,256,512) -> Wt [512][768]
__global__ void pack_wcat2_kernel(const float* __restrict__ w, bf16* __restrict__ Wt)
{
  int idx = blockIdx.x * blockDim.x + threadIdx.x;
  if (idx >= OUTD * KCAT2) return;
  const int kk = idx % KCAT2, o = idx / KCAT2;
  Wt[idx] = (bf16)w[kk * OUTD + o];
}

// rw_i (256,128,L) -> Wreg [10][256][1152], zero-padded in l
__global__ void pack_wreg_kernel(const float* __restrict__ w, int region,
                                 bf16* __restrict__ Wreg)
{
  int idx = blockIdx.x * blockDim.x + threadIdx.x;
  if (idx >= MID * KREG) return;
  const int k = idx % KREG, m = idx / KREG;
  const int l = k >> 7, c = k & 127;
  const int L = c_region_len[region];
  const float v = (l < L) ? w[(m * INCH + c) * L + l] : 0.f;
  Wreg[(long)region * MID * KREG + idx] = (bf16)v;
}

// generic fp32 (R x C) -> bf16 transpose (C x R)
__global__ void pack_transpose_kernel(const float* __restrict__ w,
                                      bf16* __restrict__ Wt, long R, long Ccols)
{
  long idx = (long)blockIdx.x * blockDim.x + threadIdx.x;
  if (idx >= R * Ccols) return;
  const long r = idx % R, c = idx / R;
  Wt[idx] = (bf16)w[r * Ccols + c];
}

// hc_w3 (256,4) -> W3t [64][256] (rows 4..63 zero); b3 -> b3pad[64]
__global__ void pack_w3_kernel(const float* __restrict__ w3,
                               const float* __restrict__ b3,
                               bf16* __restrict__ W3t, float* __restrict__ b3pad)
{
  int idx = blockIdx.x * blockDim.x + threadIdx.x;
  if (idx >= N3P * 256) return;
  const int k = idx % 256, n = idx / 256;
  W3t[idx] = (bf16)((n < 4) ? w3[k * 4 + n] : 0.f);
  if (idx < N3P) b3pad[idx] = (idx < 4) ? b3[idx] : 0.f;
}

// ---------------------------------------------------------------------------
// Batched S @ g[b] (cheby1 hop products) via WMMA: M=64, N=128, K=64.
// grid = (B, 2); block = 256 (8 waves); result -> Gcat cols [128+sel*128, +128)
// ---------------------------------------------------------------------------
__global__ __launch_bounds__(256) void chebyS_kernel(
    const bf16* __restrict__ Spad,  // [2][64][64]
    const bf16* __restrict__ gT,    // [B][128][64]
    bf16* __restrict__ Gcat)        // [B][64][384]
{
  const long b  = blockIdx.x;
  const int sel = blockIdx.y;
  const bf16* S  = Spad + sel * 64 * 64;
  const bf16* Bt = gT + b * INCH * XDP;
  const int wave = threadIdx.x >> 5, lane = threadIdx.x & 31;
  const int wm  = (wave & 3) * 16;
  const int wn0 = (wave >> 2) * 64;
  const int fr = lane & 15, fk = (lane >> 4) * 8;

  v8f acc[4] = {};
  for (int k0 = 0; k0 < 64; k0 += 32) {
    v16bf a = load_frag(&S[(wm + fr) * 64 + k0 + fk]);
#pragma unroll
    for (int c = 0; c < 4; ++c) {
      v16bf bb = load_frag(&Bt[(wn0 + c * 16 + fr) * 64 + k0 + fk]);
      acc[c] = wmma_bf16(a, bb, acc[c]);
    }
  }
  bf16* out = Gcat + b * XDP * KCAT1 + 128 + sel * 128;
  const int rbase = wm + (lane >> 4) * 8;
#pragma unroll
  for (int c = 0; c < 4; ++c) {
    const int col = wn0 + c * 16 + (lane & 15);
#pragma unroll
    for (int r = 0; r < 8; ++r)
      out[(long)(rbase + r) * KCAT1 + col] = (bf16)acc[c][r];
  }
}

// ---------------------------------------------------------------------------
// Batched A1 @ g2in[b]: M=16, N=256, K=32(pad). grid = B, block = 256.
// waves 0..3 -> A1 (cols 256..511), waves 4..7 -> A1^2 (cols 512..767)
// ---------------------------------------------------------------------------
__global__ __launch_bounds__(256) void chebyS2_kernel(
    const bf16* __restrict__ S1pad,  // [2][16][32]
    const bf16* __restrict__ g2T,    // [B][256][32]
    bf16* __restrict__ G2cat)        // [B][16][768]
{
  const long b = blockIdx.x;
  const int wave = threadIdx.x >> 5, lane = threadIdx.x & 31;
  const int sel = wave >> 2;
  const int nt0 = (wave & 3) * 4;
  const bf16* S  = S1pad + sel * 16 * 32;
  const bf16* Bt = g2T + b * MID * 32;
  const int fr = lane & 15, fk = (lane >> 4) * 8;

  v16bf a = load_frag(&S[fr * 32 + fk]);
  bf16* out = G2cat + b * K1P * KCAT2 + 256 + sel * 256;
  const int rbase = (lane >> 4) * 8;
#pragma unroll
  for (int c = 0; c < 4; ++c) {
    v16bf bb = load_frag(&Bt[((nt0 + c) * 16 + fr) * 32 + fk]);
    v8f acc = {};
    acc = wmma_bf16(a, bb, acc);
    const int col = (nt0 + c) * 16 + (lane & 15);
#pragma unroll
    for (int r = 0; r < 8; ++r)
      out[(rbase + r) * KCAT2 + col] = (bf16)acc[r];
  }
}

// ---------------------------------------------------------------------------
// Region gather: Greg[b][i][l*128+c] = g[b][idx_i[l]][c] (bf16, zero pad)
// ---------------------------------------------------------------------------
__global__ void region_gather_kernel(const bf16* __restrict__ Gcat,
                                     bf16* __restrict__ Greg)
{
  long idx = (long)blockIdx.x * blockDim.x + threadIdx.x;  // B*10*9*128
  const int c = (int)(idx & 127);
  long t = idx >> 7;
  const int l = (int)(t % 9); t /= 9;
  const int i = (int)(t % 10);
  const long b = t / 10;
  bf16 v = (bf16)0.f;
  if (l < c_region_len[i]) {
    const int ch = c_region_idx[i][l];
    v = Gcat[(b * XDP + ch) * KCAT1 + c];
  }
  Greg[idx] = v;
}

// RG (B,10,256) f32 -> G2cat cols [0,256) (rows padded to 16) + g2T (K padded 32)
__global__ void g2prep_kernel(const float* __restrict__ RG,
                              bf16* __restrict__ G2cat,
                              bf16* __restrict__ g2T)
{
  long idx = (long)blockIdx.x * blockDim.x + threadIdx.x;  // B*256
  const int m = (int)(idx & 255);
  const long b = idx >> 8;
#pragma unroll
  for (int r = 0; r < K1P; ++r) {
    const float v = (r < K1D) ? RG[(b * K1D + r) * MID + m] : 0.f;
    G2cat[(b * K1P + r) * KCAT2 + m] = (bf16)v;
  }
#pragma unroll
  for (int r = 0; r < 32; ++r) {
    const float v = (r < K1D) ? RG[(b * K1D + r) * MID + m] : 0.f;
    g2T[(b * MID + m) * 32 + r] = (bf16)v;
  }
}

// ---------------------------------------------------------------------------
// Gated softmax fusion over j=0..71 per (b,o); writes bf16 (B,36864)
// ---------------------------------------------------------------------------
__global__ void softmax_gather_kernel(const float* __restrict__ g1,  // [B][64][512]
                                      const float* __restrict__ g2,  // [B][16][512]
                                      const float* __restrict__ bg,
                                      const float* __restrict__ cg,
                                      bf16* __restrict__ Aout)       // [B][36864]
{
  long idx = (long)blockIdx.x * blockDim.x + threadIdx.x;  // B*512
  const int o = (int)(idx & 511);
  const long b = idx >> 9;
  const float bgate = bg[0], cgate = cg[0];
  const float* p1 = g1 + (b * XDP) * OUTD + o;
  const float* p2 = g2 + (b * K1P) * OUTD + o;

  float m = -3.4e38f, s = 0.f;
  for (int j = 0; j < 72; ++j) {
    const float v = (j < XD) ? p1[j * OUTD] : p2[(j - XD) * OUTD];
    const float a = ((j < XD) ? bgate : cgate) * v;
    if (a > m) { s = s * __expf(m - a) + 1.f; m = a; }
    else       { s += __expf(a - m); }
  }
  const float inv = 1.f / s;
  bf16* op = Aout + b * FEAT + o;
  for (int j = 0; j < 72; ++j) {
    const float v = (j < XD) ? p1[j * OUTD] : p2[(j - XD) * OUTD];
    const float a = ((j < XD) ? bgate : cgate) * v;
    op[j * OUTD] = (bf16)(__expf(a - m) * inv * v);
  }
}

// ---------------------------------------------------------------------------
// BatchNorm stats (population var) + apply + exact GELU -> bf16
// ---------------------------------------------------------------------------
__global__ void bn_stats_kernel(const float* __restrict__ H, int rows, int cols,
                                float* __restrict__ mu, float* __restrict__ var)
{
  const int c = blockIdx.x;
  __shared__ float s1[256], s2[256];
  float a = 0.f, q = 0.f;
  for (int r = threadIdx.x; r < rows; r += 256) {
    const float v = H[(long)r * cols + c];
    a += v; q += v * v;
  }
  s1[threadIdx.x] = a; s2[threadIdx.x] = q;
  __syncthreads();
  for (int st = 128; st > 0; st >>= 1) {
    if ((int)threadIdx.x < st) {
      s1[threadIdx.x] += s1[threadIdx.x + st];
      s2[threadIdx.x] += s2[threadIdx.x + st];
    }
    __syncthreads();
  }
  if (threadIdx.x == 0) {
    const float mean = s1[0] / rows;
    mu[c]  = mean;
    var[c] = s2[0] / rows - mean * mean;
  }
}

__global__ void bn_gelu_kernel(const float* __restrict__ H,
                               const float* __restrict__ mu,
                               const float* __restrict__ var,
                               const float* __restrict__ gam,
                               const float* __restrict__ bet,
                               bf16* __restrict__ Out, long elems, int cols)
{
  long idx = (long)blockIdx.x * blockDim.x + threadIdx.x;
  if (idx >= elems) return;
  const int c = (int)(idx % cols);
  const float xv = H[idx];
  const float xn = (xv - mu[c]) * rsqrtf(var[c] + 1e-5f) * gam[c] + bet[c];
  Out[idx] = (bf16)(0.5f * xn * (1.f + erff(xn * 0.70710678118654752f)));
}

__global__ void softmax4_kernel(const float* __restrict__ H3,  // [B][64]
                                float* __restrict__ out)       // [B][4]
{
  const int b = blockIdx.x * blockDim.x + threadIdx.x;
  if (b >= NB) return;
  const float v0 = H3[b * N3P + 0], v1 = H3[b * N3P + 1];
  const float v2 = H3[b * N3P + 2], v3 = H3[b * N3P + 3];
  const float m = fmaxf(fmaxf(v0, v1), fmaxf(v2, v3));
  const float e0 = __expf(v0 - m), e1 = __expf(v1 - m);
  const float e2 = __expf(v2 - m), e3 = __expf(v3 - m);
  const float inv = 1.f / (e0 + e1 + e2 + e3);
  out[b * 4 + 0] = e0 * inv; out[b * 4 + 1] = e1 * inv;
  out[b * 4 + 2] = e2 * inv; out[b * 4 + 3] = e3 * inv;
}

// ---------------------------------------------------------------------------
// Host-side orchestration
// ---------------------------------------------------------------------------
extern "C" void kernel_launch(void* const* d_in, const int* in_sizes, int n_in,
                              void* d_out, int out_size, void* d_ws, size_t ws_size,
                              hipStream_t stream)
{
  const float* x       = (const float*)d_in[0];
  const float* tfe_w   = (const float*)d_in[1];
  const float* tfe_b   = (const float*)d_in[2];
  const float* cheby1w = (const float*)d_in[3];
  const float* Amat    = (const float*)d_in[4];
  const float* cheby2w = (const float*)d_in[5];
  const float* A1mat   = (const float*)d_in[6];
  const float* b_gate  = (const float*)d_in[7];
  const float* c_gate  = (const float*)d_in[8];
  const float* region_b= (const float*)d_in[9];
  const float* hc_w1   = (const float*)d_in[10];
  const float* hc_b1   = (const float*)d_in[11];
  const float* bn1_g   = (const float*)d_in[12];
  const float* bn1_b   = (const float*)d_in[13];
  const float* hc_w2   = (const float*)d_in[14];
  const float* hc_b2   = (const float*)d_in[15];
  const float* bn2_g   = (const float*)d_in[16];
  const float* bn2_b   = (const float*)d_in[17];
  const float* hc_w3   = (const float*)d_in[18];
  const float* hc_b3   = (const float*)d_in[19];
  const float* rw[10];
  for (int i = 0; i < 10; ++i) rw[i] = (const float*)d_in[20 + i];
  float* outp = (float*)d_out;

  // --- workspace bump allocator ---
  char* wsb = (char*)d_ws; size_t off = 0;
  auto alloc = [&](size_t bytes) -> void* {
    void* p = wsb + off;
    off = (off + bytes + 255) & ~(size_t)255;
    return p;
  };

  bf16* Spad   = (bf16*)alloc(2 * 64 * 64 * sizeof(bf16));
  float* A2tmp = (float*)alloc(XD * XD * sizeof(float));
  bf16* S1pad  = (bf16*)alloc(2 * 16 * 32 * sizeof(bf16));
  float* A1sq  = (float*)alloc(100 * sizeof(float));
  bf16* Wcat1  = (bf16*)alloc((size_t)OUTD * KCAT1 * sizeof(bf16));
  bf16* Wreg   = (bf16*)alloc((size_t)10 * MID * KREG * sizeof(bf16));
  bf16* Wcat2  = (bf16*)alloc((size_t)OUTD * KCAT2 * sizeof(bf16));
  bf16* W1t    = (bf16*)alloc((size_t)LIN * FEAT * sizeof(bf16));
  bf16* W2t    = (bf16*)alloc((size_t)256 * 512 * sizeof(bf16));
  bf16* W3t    = (bf16*)alloc((size_t)N3P * 256 * sizeof(bf16));
  float* b3pad = (float*)alloc(N3P * sizeof(float));
  bf16* Gcat   = (bf16*)alloc((size_t)NB * XDP * KCAT1 * sizeof(bf16));
  bf16* gT     = (bf16*)alloc((size_t)NB * INCH * XDP * sizeof(bf16));
  bf16* Greg   = (bf16*)alloc((size_t)NB * 10 * KREG * sizeof(bf16));
  float* RG    = (float*)alloc((size_t)NB * K1D * MID * sizeof(float));
  bf16* G2cat  = (bf16*)alloc((size_t)NB * K1P * KCAT2 * sizeof(bf16));
  bf16* g2T    = (bf16*)alloc((size_t)NB * MID * 32 * sizeof(bf16));
  float* g1    = (float*)alloc((size_t)NB * XDP * OUTD * sizeof(float));
  float* g2    = (float*)alloc((size_t)NB * K1P * OUTD * sizeof(float));
  bf16* Aout   = (bf16*)alloc((size_t)NB * FEAT * sizeof(bf16));
  float* Cpart = (float*)alloc((size_t)16 * NB * LIN * sizeof(float));
  float* H1    = (float*)alloc((size_t)NB * LIN * sizeof(float));
  float* mu1   = (float*)alloc(LIN * sizeof(float));
  float* var1  = (float*)alloc(LIN * sizeof(float));
  bf16* H1g    = (bf16*)alloc((size_t)NB * LIN * sizeof(bf16));
  float* H2    = (float*)alloc((size_t)NB * 256 * sizeof(float));
  float* mu2   = (float*)alloc(256 * sizeof(float));
  float* var2  = (float*)alloc(256 * sizeof(float));
  bf16* H2g    = (bf16*)alloc((size_t)NB * 256 * sizeof(bf16));
  float* H3    = (float*)alloc((size_t)NB * N3P * sizeof(float));
  (void)ws_size; (void)in_sizes; (void)n_in; (void)out_size;

  const int T = 256;

  // --- weight / operator packing (deterministic, every call) ---
  matmul62_kernel<<<(XD * XD + T - 1) / T, T, 0, stream>>>(Amat, A2tmp);
  pack_S_kernel<<<(2 * 64 * 64) / T, T, 0, stream>>>(Amat, A2tmp, Spad);
  matmul10_kernel<<<1, T, 0, stream>>>(A1mat, A1sq);
  pack_S1_kernel<<<(2 * 16 * 32 + T - 1) / T, T, 0, stream>>>(A1mat, A1sq, S1pad);
  pack_wcat1_kernel<<<(OUTD * KCAT1) / T, T, 0, stream>>>(cheby1w, Wcat1);
  pack_wcat2_kernel<<<(OUTD * KCAT2) / T, T, 0, stream>>>(cheby2w, Wcat2);
  for (int i = 0; i < 10; ++i)
    pack_wreg_kernel<<<(MID * KREG) / T, T, 0, stream>>>(rw[i], i, Wreg);
  pack_transpose_kernel<<<((long)FEAT * LIN) / T, T, 0, stream>>>(hc_w1, W1t, FEAT, LIN);
  pack_transpose_kernel<<<(512 * 256) / T, T, 0, stream>>>(hc_w2, W2t, 512, 256);
  pack_w3_kernel<<<(N3P * 256) / T, T, 0, stream>>>(hc_w3, hc_b3, W3t, b3pad);

  // --- stage 1: preproc -> g (bf16, two layouts) ---
  preproc_kernel<<<(NB * XDP * INCH) / T, T, 0, stream>>>(x, tfe_w, tfe_b, Gcat, gT);

  // --- cheby1 hop products: Gcat[:,:,128:384] = [A g | A^2 g] ---
  chebyS_kernel<<<dim3(NB, 2, 1), T, 0, stream>>>(Spad, gT, Gcat);

  // --- cheby1 GEMM: (B*64, 384) x (384, 512) -> relu -> g1 ---
  gemm_bf16_kernel<<<dim3(NB * XDP / BM, OUTD / BN, 1), T, 0, stream>>>(
      Gcat, KCAT1, 0, Wcat1, KCAT1, 0, g1, OUTD, 0,
      NB * XDP, KCAT1, nullptr, 0, 1, 1);

  // --- region pooling: 10 batched GEMMs (grid.z = region) ---
  region_gather_kernel<<<((long)NB * 10 * KREG) / T, T, 0, stream>>>(Gcat, Greg);
  gemm_bf16_kernel<<<dim3(NB / BM, MID / BN, 10), T, 0, stream>>>(
      Greg, (long)10 * KREG, KREG,
      Wreg, KREG, (long)MID * KREG,
      RG, (long)K1D * MID, MID,
      NB, KREG, region_b, MID, 0, 1);

  // --- cheby2: pad/convert, hop products, GEMM -> relu -> g2 ---
  g2prep_kernel<<<(NB * MID) / T, T, 0, stream>>>(RG, G2cat, g2T);
  chebyS2_kernel<<<dim3(NB, 1, 1), T, 0, stream>>>(S1pad, g2T, G2cat);
  gemm_bf16_kernel<<<dim3(NB * K1P / BM, OUTD / BN, 1), T, 0, stream>>>(
      G2cat, KCAT2, 0, Wcat2, KCAT2, 0, g2, OUTD, 0,
      NB * K1P, KCAT2, nullptr, 0, 1, 1);

  // --- gated softmax fusion -> Aout (B, 36864) bf16 ---
  softmax_gather_kernel<<<(NB * OUTD) / T, T, 0, stream>>>(g1, g2, b_gate, c_gate, Aout);

  // --- head layer 1: split-K=16 GEMM + deterministic reduce + bias ---
  gemm_bf16_kernel<<<dim3(NB / BM, LIN / BN, 16), T, 0, stream>>>(
      Aout, FEAT, 0, W1t, FEAT, 0, Cpart, LIN, 0,
      NB, FEAT, nullptr, 0, 0, 16);
  ksplit_reduce_kernel<<<((long)NB * LIN) / T, T, 0, stream>>>(
      Cpart, H1, hc_b1, (long)NB * LIN, LIN, 16);
  bn_stats_kernel<<<LIN, T, 0, stream>>>(H1, NB, LIN, mu1, var1);
  bn_gelu_kernel<<<((long)NB * LIN) / T, T, 0, stream>>>(
      H1, mu1, var1, bn1_g, bn1_b, H1g, (long)NB * LIN, LIN);

  // --- head layer 2 ---
  gemm_bf16_kernel<<<dim3(NB / BM, 256 / BN, 1), T, 0, stream>>>(
      H1g, LIN, 0, W2t, LIN, 0, H2, 256, 0,
      NB, LIN, hc_b2, 0, 0, 1);
  bn_stats_kernel<<<256, T, 0, stream>>>(H2, NB, 256, mu2, var2);
  bn_gelu_kernel<<<((long)NB * 256) / T, T, 0, stream>>>(
      H2, mu2, var2, bn2_g, bn2_b, H2g, (long)NB * 256, 256);

  // --- head layer 3 (N padded 4->64, guard-free) + final softmax ---
  gemm_bf16_kernel<<<dim3(NB / BM, N3P / BN, 1), T, 0, stream>>>(
      H2g, 256, 0, W3t, 256, 0, H3, N3P, 0,
      NB, 256, b3pad, 0, 0, 1);
  softmax4_kernel<<<(NB + T - 1) / T, T, 0, stream>>>(H3, outp);
}
